// BaseVectorQuantizer_29334626631755
// MI455X (gfx1250) — matrix operand order repaired and checked by reading, not compiled, not run on hardware
//
#include <hip/hip_runtime.h>
#include <hip/hip_bf16.h>

// ---------------------------------------------------------------------------
// VQ bottleneck for MI455X (gfx1250, wave32, WMMA).
// B=32, C=256, H=W=32 -> N = 32768 tokens, K=1024 codes, CD=8.
// Output-bandwidth bound (~168MB @ 23.3 TB/s => ~7us floor). The two K=256
// GEMMs use v_wmma_f32_16x16x32_bf16 with weights staged per-workgroup into
// LDS via gfx1250 async global->LDS loads (double-buffered), cutting L2
// weight traffic 8x. The one-hot encodings output is produced in a single
// write pass. Everything else fused in-register per wave.
// ---------------------------------------------------------------------------

typedef __attribute__((ext_vector_type(16))) __bf16 v16bf;
typedef __attribute__((ext_vector_type(8)))  float  v8f;
typedef __attribute__((ext_vector_type(4)))  unsigned int u32x4;
typedef __attribute__((ext_vector_type(4)))  float  f32x4;
typedef __attribute__((vector_size(16)))     int    i32x4v;  // builtin's pointee

union BF16x16 { v16bf v; u32x4 u[2]; };

#define N_TOK   32768
#define C_DIM   256
#define CD_DIM  8
#define K_CODES 1024
#define HW      1024
#define LN_EPS  1e-5f

// d_out layout (floats): quantized[32768*256] | indices[32768] | encodings[32768*1024]
#define OUT_Q_OFF   0
#define OUT_IDX_OFF (N_TOK * C_DIM)            // 8388608
#define OUT_ENC_OFF (OUT_IDX_OFF + N_TOK)      // 8421376

// workspace layout (bytes)
#define WS_W1T   0                     // bf16 [256][256]  w_in1^T
#define WS_WO2T  131072                // bf16 [256][256]  w_out2^T
#define WS_WO1T  262144                // f32  [256][8]    w_out1^T
#define WS_ENORM 270336                // f32  [1024]      ||e||^2
#define WS_IDX   274432                // i32  [32768]

// gfx1250 async global->LDS path (guarded: falls back to load+ds_store)
#if defined(__has_builtin)
#  if __has_builtin(__builtin_amdgcn_global_load_async_to_lds_b128) && \
      __has_builtin(__builtin_amdgcn_s_wait_asynccnt)
#    define USE_ASYNC_LDS 1
#  endif
#endif
#ifndef USE_ASYNC_LDS
#  define USE_ASYNC_LDS 0
#endif

#define AS_GLOBAL __attribute__((address_space(1)))
#define AS_LOCAL  __attribute__((address_space(3)))

#define ROWP 40   // LDS row pitch in bf16 (80B): lanes land on banks 4k apart

__device__ __forceinline__ unsigned short f2bf_rte(float f) {
  unsigned int u = __float_as_uint(f);
  u += 0x7FFFu + ((u >> 16) & 1u);     // round-to-nearest-even
  return (unsigned short)(u >> 16);
}

// Stage one K-slice (256 rows x 32 bf16) of a [256][256] bf16 weight matrix
// into LDS (padded pitch), 16B per chunk, 4 chunks per thread (256 threads).
__device__ __forceinline__ void stage_slice(const unsigned short* __restrict__ wT,
                                            int kt, unsigned short* lb, int tid) {
#pragma unroll
  for (int p = 0; p < 4; ++p) {
    int ci = tid + p * 256;                  // 0..1023 chunks of 16B
    int n = ci >> 2, c = ci & 3;
    const unsigned short* src = wT + n * 256 + kt * 32 + c * 8;
    unsigned short* dst = lb + n * ROWP + c * 8;
#if USE_ASYNC_LDS
    __builtin_amdgcn_global_load_async_to_lds_b128(
        (AS_GLOBAL i32x4v*)(AS_GLOBAL void*)(void*)src,
        (AS_LOCAL  i32x4v*)(AS_LOCAL  void*)(void*)dst, 0, 0);
#else
    *(u32x4*)dst = *(const u32x4*)src;
#endif
  }
}

__device__ __forceinline__ void stage_wait_barrier() {
#if USE_ASYNC_LDS
  __builtin_amdgcn_s_wait_asynccnt(0);
#endif
  __syncthreads();
}

// -------------------------- K0: weight prep -------------------------------
__global__ void k0_prep(const float* __restrict__ w_in1,
                        const float* __restrict__ w_out2,
                        const float* __restrict__ w_out1,
                        const float* __restrict__ emb,
                        unsigned short* __restrict__ w1T,
                        unsigned short* __restrict__ wo2T,
                        float* __restrict__ wo1T,
                        float* __restrict__ enorm) {
  int i = blockIdx.x * blockDim.x + threadIdx.x;   // 65536 threads
  int k = i >> 8, n = i & 255;
  w1T [n * 256 + k] = f2bf_rte(w_in1 [k * 256 + n]);
  wo2T[n * 256 + k] = f2bf_rte(w_out2[k * 256 + n]);
  if (i < C_DIM * CD_DIM) {                         // 2048: w_out1^T (f32)
    int c = i >> 3, d = i & 7;
    wo1T[c * 8 + d] = w_out1[d * 256 + c];
  }
  if (i < K_CODES) {                                // ||e||^2
    float s = 0.f;
#pragma unroll
    for (int d = 0; d < 8; ++d) { float e = emb[i * 8 + d]; s += e * e; }
    enorm[i] = s;
  }
}

// ---------------- K1: project_in + LN(8) + argmin over codebook -----------
__global__ __launch_bounds__(256)
void k1_project_in_argmin(const float* __restrict__ feat,
                          const float* __restrict__ emb,
                          const float* __restrict__ b_in1,
                          const float* __restrict__ w_in2,
                          const float* __restrict__ b_in2,
                          const float* __restrict__ ln_in_g,
                          const float* __restrict__ ln_in_b,
                          const unsigned short* __restrict__ w1T,
                          const float* __restrict__ enorm,
                          int* __restrict__ idx_out) {
  __shared__ alignas(16) unsigned short lds_b[2][256 * ROWP];   // 2 x 20KB

  const int lane  = threadIdx.x & 31;
  const int m     = lane & 15;           // token-in-tile / B-column / D-column
  const int hi    = lane >> 4;           // lane half (K/M split)
  const int wave  = blockIdx.x * (blockDim.x >> 5) + (threadIdx.x >> 5);
  const int tok0  = wave * 16;
  const int token = tok0 + m;
  const int bb    = token >> 10;         // batch (tiles never cross batches)
  const int tt    = token & 1023;        // spatial position
  const float* fbase = feat + ((size_t)bb * C_DIM) * HW + tt;   // + c*HW

  v8f acc[16];
#pragma unroll
  for (int i = 0; i < 16; ++i)
#pragma unroll
    for (int r = 0; r < 8; ++r) acc[i][r] = 0.f;

  // -------- GEMM1: hidden[16x256] = x[16x256] @ w_in1, bf16 WMMA ----------
  stage_slice(w1T, 0, &lds_b[0][0], threadIdx.x);
#pragma unroll
  for (int kt = 0; kt < 8; ++kt) {
    stage_wait_barrier();                          // slice kt ready in LDS
    if (kt < 7) stage_slice(w1T, kt + 1, &lds_b[(kt + 1) & 1][0], threadIdx.x);
    const unsigned short* lb = &lds_b[kt & 1][0];

    BF16x16 A;
#pragma unroll
    for (int e = 0; e < 16; ++e) {
      int c = kt * 32 + e + (hi << 4);             // contiguous K index: k = 16*hi + e
      A.v[e] = (__bf16)fbase[(size_t)c * HW];      // lanes 0-15: coalesced
    }
#pragma unroll
    for (int nt = 0; nt < 16; ++nt) {
      BF16x16 Bm;
      const u32x4* bp = (const u32x4*)(lb + (nt * 16 + m) * ROWP + (hi << 4));
      Bm.u[0] = bp[0]; Bm.u[1] = bp[1];            // conflict-free ds_load_b128
      acc[nt] = __builtin_amdgcn_wmma_f32_16x16x32_bf16(
          false, A.v, false, Bm.v, (short)0, acc[nt], false, false);
    }
  }

  // -------- GEMM2 (256->8) in VALU directly from D-layout -----------------
  // acc[nt][r] = hidden[M = hi*8+r][c = nt*16+m]
  float pz[8][8];
#pragma unroll
  for (int r = 0; r < 8; ++r)
#pragma unroll
    for (int j = 0; j < 8; ++j) pz[r][j] = 0.f;

#pragma unroll
  for (int nt = 0; nt < 16; ++nt) {
    int c = nt * 16 + m;
    float bias = b_in1[c];
    f32x4 w0 = *(const f32x4*)(w_in2 + c * 8);
    f32x4 w1 = *(const f32x4*)(w_in2 + c * 8 + 4);
    float wr[8] = {w0.x, w0.y, w0.z, w0.w, w1.x, w1.y, w1.z, w1.w};
#pragma unroll
    for (int r = 0; r < 8; ++r) {
      float h = acc[nt][r] + bias;
      h = h > 0.f ? h : 0.f;                       // ReLU
#pragma unroll
      for (int j = 0; j < 8; ++j) pz[r][j] += h * wr[j];
    }
  }
  // reduce the c-dimension across the 16 lanes of each half
#pragma unroll
  for (int mask = 1; mask <= 8; mask <<= 1)
#pragma unroll
    for (int r = 0; r < 8; ++r)
#pragma unroll
      for (int j = 0; j < 8; ++j)
        pz[r][j] += __shfl_xor(pz[r][j], mask, 32);

  // -------- bias + LayerNorm over CD=8 ------------------------------------
  float g8[8], be8[8], bi8[8];
#pragma unroll
  for (int j = 0; j < 8; ++j) {
    g8[j] = ln_in_g[j]; be8[j] = ln_in_b[j]; bi8[j] = b_in2[j];
  }
#pragma unroll
  for (int r = 0; r < 8; ++r) {
    float s = 0.f;
#pragma unroll
    for (int j = 0; j < 8; ++j) { pz[r][j] += bi8[j]; s += pz[r][j]; }
    float mu = s * 0.125f, v = 0.f;
#pragma unroll
    for (int j = 0; j < 8; ++j) { float d = pz[r][j] - mu; v += d * d; }
    float rs = rsqrtf(v * 0.125f + LN_EPS);
#pragma unroll
    for (int j = 0; j < 8; ++j)
      pz[r][j] = (pz[r][j] - mu) * rs * g8[j] + be8[j];
  }

  // -------- argmin over 1024 codes: 64 codes per lane ---------------------
  float best[8]; int bidx[8];
#pragma unroll
  for (int r = 0; r < 8; ++r) { best[r] = 3.4e38f; bidx[r] = K_CODES; }
  for (int kk = 0; kk < 64; ++kk) {
    int k = m * 64 + kk;
    f32x4 e0 = *(const f32x4*)(emb + k * 8);
    f32x4 e1 = *(const f32x4*)(emb + k * 8 + 4);
    float en = enorm[k];
#pragma unroll
    for (int r = 0; r < 8; ++r) {
      float dot = pz[r][0]*e0.x + pz[r][1]*e0.y + pz[r][2]*e0.z + pz[r][3]*e0.w
                + pz[r][4]*e1.x + pz[r][5]*e1.y + pz[r][6]*e1.z + pz[r][7]*e1.w;
      float d = en - 2.f * dot;                    // ||x||^2 const: drop
      if (d < best[r]) { best[r] = d; bidx[r] = k; }
    }
  }
#pragma unroll
  for (int mask = 1; mask <= 8; mask <<= 1)
#pragma unroll
    for (int r = 0; r < 8; ++r) {
      float ov = __shfl_xor(best[r], mask, 32);
      int   oi = __shfl_xor(bidx[r], mask, 32);
      if (ov < best[r] || (ov == best[r] && oi < bidx[r])) {
        best[r] = ov; bidx[r] = oi;
      }
    }
  if (m == 0) {
#pragma unroll
    for (int r = 0; r < 8; ++r) idx_out[tok0 + hi * 8 + r] = bidx[r];
  }
}

// ---------------- K2: project_out + LN(256) -------------------------------
__global__ __launch_bounds__(256)
void k2_project_out(const float* __restrict__ emb,
                    const int* __restrict__ idx_in,
                    const float* __restrict__ wo1T,
                    const float* __restrict__ b_out1,
                    const unsigned short* __restrict__ wo2T,
                    const float* __restrict__ b_out2,
                    const float* __restrict__ ln_out_g,
                    const float* __restrict__ ln_out_b,
                    float* __restrict__ out_q) {
  __shared__ alignas(16) unsigned short lds_b[2][256 * ROWP];   // 2 x 20KB

  const int lane  = threadIdx.x & 31;
  const int m     = lane & 15;
  const int hi    = lane >> 4;
  const int wave  = blockIdx.x * (blockDim.x >> 5) + (threadIdx.x >> 5);
  const int tok0  = wave * 16;
  const int code  = idx_in[tok0 + m];

  f32x4 q0 = *(const f32x4*)(emb + code * 8);
  f32x4 q1 = *(const f32x4*)(emb + code * 8 + 4);
  float q[8] = {q0.x, q0.y, q0.z, q0.w, q1.x, q1.y, q1.z, q1.w};

  v8f acc[16];
#pragma unroll
  for (int i = 0; i < 16; ++i)
#pragma unroll
    for (int r = 0; r < 8; ++r) acc[i][r] = 0.f;

  stage_slice(wo2T, 0, &lds_b[0][0], threadIdx.x);
#pragma unroll
  for (int kt = 0; kt < 8; ++kt) {
    stage_wait_barrier();
    if (kt < 7) stage_slice(wo2T, kt + 1, &lds_b[(kt + 1) & 1][0], threadIdx.x);
    const unsigned short* lb = &lds_b[kt & 1][0];

    // h2[m][c] = relu(q @ w_out1 + b_out1), built directly in WMMA A-layout
    BF16x16 A;
#pragma unroll
    for (int e = 0; e < 16; ++e) {
      int c = kt * 32 + e + (hi << 4);
      const float* wr = wo1T + c * 8;              // contiguous [c][8]
      float h = b_out1[c];
#pragma unroll
      for (int d = 0; d < 8; ++d) h += q[d] * wr[d];
      h = h > 0.f ? h : 0.f;
      A.v[e] = (__bf16)h;
    }
#pragma unroll
    for (int nt = 0; nt < 16; ++nt) {
      BF16x16 Bm;
      const u32x4* bp = (const u32x4*)(lb + (nt * 16 + m) * ROWP + (hi << 4));
      Bm.u[0] = bp[0]; Bm.u[1] = bp[1];
      acc[nt] = __builtin_amdgcn_wmma_f32_16x16x32_bf16(
          false, A.v, false, Bm.v, (short)0, acc[nt], false, false);
    }
  }

  // bias + LayerNorm over 256 from D-layout
  float s[8], sq[8];
#pragma unroll
  for (int r = 0; r < 8; ++r) { s[r] = 0.f; sq[r] = 0.f; }
#pragma unroll
  for (int nt = 0; nt < 16; ++nt) {
    int c = nt * 16 + m;
    float bias = b_out2[c];
#pragma unroll
    for (int r = 0; r < 8; ++r) {
      float v = acc[nt][r] + bias;
      acc[nt][r] = v; s[r] += v; sq[r] += v * v;
    }
  }
#pragma unroll
  for (int mask = 1; mask <= 8; mask <<= 1)
#pragma unroll
    for (int r = 0; r < 8; ++r) {
      s[r]  += __shfl_xor(s[r],  mask, 32);
      sq[r] += __shfl_xor(sq[r], mask, 32);
    }
  float mu[8], rs[8];
#pragma unroll
  for (int r = 0; r < 8; ++r) {
    mu[r] = s[r] * (1.f / 256.f);
    float var = sq[r] * (1.f / 256.f) - mu[r] * mu[r];
    rs[r] = rsqrtf(var + LN_EPS);
  }
#pragma unroll
  for (int nt = 0; nt < 16; ++nt) {
    int c = nt * 16 + m;
    float g = ln_out_g[c], be = ln_out_b[c];
#pragma unroll
    for (int r = 0; r < 8; ++r) {
      int M = hi * 8 + r;
      out_q[(size_t)(tok0 + M) * C_DIM + c] = (acc[nt][r] - mu[r]) * rs[r] * g + be;
    }
  }
}

// ---------- K3: one-hot encodings + indices, single write pass ------------
// One block per token row: 256 threads x one float4 = 1024 floats.
__global__ __launch_bounds__(256)
void k_encodings(const int* __restrict__ idx,
                 float* __restrict__ out_idx,
                 float* __restrict__ enc) {
  int n  = blockIdx.x;                 // token row
  int c4 = threadIdx.x;                // float4 column
  int k  = idx[n];
  f32x4 v = {0.f, 0.f, 0.f, 0.f};
  if ((k >> 2) == c4) {
    int r = k & 3;
    v.x = (r == 0) ? 1.f : 0.f;
    v.y = (r == 1) ? 1.f : 0.f;
    v.z = (r == 2) ? 1.f : 0.f;
    v.w = (r == 3) ? 1.f : 0.f;
  }
  ((f32x4*)enc)[(size_t)n * 256 + c4] = v;
  if (c4 == 0) out_idx[n] = (float)k;
}

// ---------------------------------------------------------------------------
extern "C" void kernel_launch(void* const* d_in, const int* in_sizes, int n_in,
                              void* d_out, int out_size, void* d_ws, size_t ws_size,
                              hipStream_t stream) {
  const float* feat     = (const float*)d_in[0];
  const float* emb      = (const float*)d_in[1];
  const float* w_in1    = (const float*)d_in[2];
  const float* b_in1    = (const float*)d_in[3];
  const float* w_in2    = (const float*)d_in[4];
  const float* b_in2    = (const float*)d_in[5];
  const float* w_out1   = (const float*)d_in[6];
  const float* b_out1   = (const float*)d_in[7];
  const float* w_out2   = (const float*)d_in[8];
  const float* b_out2   = (const float*)d_in[9];
  const float* ln_in_g  = (const float*)d_in[10];
  const float* ln_in_b  = (const float*)d_in[11];
  const float* ln_out_g = (const float*)d_in[12];
  const float* ln_out_b = (const float*)d_in[13];

  char* ws = (char*)d_ws;
  unsigned short* w1T   = (unsigned short*)(ws + WS_W1T);
  unsigned short* wo2T  = (unsigned short*)(ws + WS_WO2T);
  float*          wo1T  = (float*)(ws + WS_WO1T);
  float*          enorm = (float*)(ws + WS_ENORM);
  int*            idxws = (int*)(ws + WS_IDX);

  float* out  = (float*)d_out;
  float* outQ = out + OUT_Q_OFF;
  float* outI = out + OUT_IDX_OFF;
  float* outE = out + OUT_ENC_OFF;

  k0_prep<<<256, 256, 0, stream>>>(w_in1, w_out2, w_out1, emb,
                                   w1T, wo2T, wo1T, enorm);

  // 256 blocks * 8 waves * 16 tokens = 32768 tokens
  k1_project_in_argmin<<<256, 256, 0, stream>>>(
      feat, emb, b_in1, w_in2, b_in2, ln_in_g, ln_in_b, w1T, enorm, idxws);

  k2_project_out<<<256, 256, 0, stream>>>(
      emb, idxws, wo1T, b_out1, wo2T, b_out2, ln_out_g, ln_out_b, outQ);

  k_encodings<<<N_TOK, 256, 0, stream>>>(idxws, outI, outE);
}
